// GraphAttentionV2Layer_48120813584667
// MI455X (gfx1250) — matrix operand order repaired and checked by reading, not compiled, not run on hardware
//
#include <hip/hip_runtime.h>
#include <hip/hip_bf16.h>

// ---------------------------------------------------------------------------
// GATv2 layer for MI455X (gfx1250, wave32, WMMA bf16 16x16x32 + TDM)
//   N=1024 nodes, F=256 in-features, H=8 heads, D=16 hidden (out = H*D = 128)
// Pipeline:
//   K1: projections g_l = h@W_l, g_r = h@W_r via v_wmma_f32_16x16x32_bf16,
//       W tiles DMA'd into LDS by the Tensor Data Mover (tensor_load_to_lds)
//   K2: fused flash-style masked softmax + (P @ g_r) aggregation; P.V on WMMA,
//       V tiles DMA'd into LDS by TDM, next g_l block prefetched.
// ---------------------------------------------------------------------------

constexpr int N_NODES = 1024;
constexpr int F_IN    = 256;
constexpr int H_HEADS = 8;
constexpr int D_HID   = 16;
constexpr int HD      = H_HEADS * D_HID;   // 128
#define SLOPE   0.2f
#define NEG_BIG (-1.0e30f)

typedef __attribute__((ext_vector_type(16))) __bf16       v16bf;
typedef __attribute__((ext_vector_type(8)))  float        v8f;
typedef __attribute__((ext_vector_type(4)))  unsigned int v4u;
typedef __attribute__((ext_vector_type(4)))  int          v4i;
typedef __attribute__((ext_vector_type(8)))  int          v8i;

// Compiler-level fence: the TDM builtin's LDS writes are not in clang's alias
// model, so LDS reads around it must not be CSE'd/hoisted across the DMA.
#define TDM_COMPILER_FENCE() asm volatile("" ::: "memory")

// ---------------------------------------------------------------------------
// Issue a TDM 2-D tile load Global -> LDS (CDNA5 ISA ch.8 D# layout).
//   data_size_log: 0=1B, 1=2B, 2=4B       tile is tile_d1 rows x tile_d0 elems
//   stride0: tensor row stride in elements
// This toolchain's builtin takes 6 args (clang-23 / therock-10.0 form):
//   (v4u group0, v8i group1, v4i group2, v4i group3, v8i extra, i32 cpol)
// ---------------------------------------------------------------------------
__device__ __forceinline__ void tdm_load_2d(unsigned lds_off, const void* gptr,
                                            unsigned data_size_log,
                                            unsigned tensor_d0, unsigned tensor_d1,
                                            unsigned tile_d0,   unsigned tile_d1,
                                            unsigned stride0)
{
    const unsigned long long ga = (unsigned long long)(uintptr_t)gptr;
    // group0: [1:0]=count=1 | [63:32]=lds_addr | [120:64]=global_addr | [127:126]=2
    v4u g0 = { 1u, lds_off, (unsigned)ga,
               (unsigned)((ga >> 32) & 0x01FFFFFFu) | 0x80000000u };
    // group1: [17:16]=data_size | [79:48]=tensor_dim0 | [111:80]=tensor_dim1 |
    //         [127:112]=tile_dim0 | [143:128]=tile_dim1 | [207:160]=dim0_stride
    v8i g1 = { (int)(data_size_log << 16),
               (int)((tensor_d0 & 0xFFFFu) << 16),
               (int)(((tensor_d0 >> 16) & 0xFFFFu) | ((tensor_d1 & 0xFFFFu) << 16)),
               (int)(((tensor_d1 >> 16) & 0xFFFFu) | ((tile_d0 & 0xFFFFu) << 16)),
               (int)(tile_d1 & 0xFFFFu),
               (int)stride0, 0, 0 };
    v4i gz4 = { 0, 0, 0, 0 };
    v8i gz8 = { 0, 0, 0, 0, 0, 0, 0, 0 };
    TDM_COMPILER_FENCE();          // don't sink pending LDS reads past the DMA
    __builtin_amdgcn_tensor_load_to_lds(g0, g1, gz4, gz4, gz8, 0);
}

// Wait for all TDM transfers, then force the compiler to re-read LDS.
__device__ __forceinline__ void tdm_wait_all()
{
    __builtin_amdgcn_s_wait_tensorcnt(0);
    TDM_COMPILER_FENCE();
}

// ---------------------------------------------------------------------------
// Kernel 1: per-wave 16x16 output tile of G = h @ W  (one col-tile == one head)
// grid (N/16, H, 2): z=0 -> W_l -> g_l ; z=1 -> W_r -> g_r (+ bf16 copy for K2)
// ---------------------------------------------------------------------------
__global__ __launch_bounds__(32)
void gat_proj_kernel(const float* __restrict__ h,      // [N, F]
                     const float* __restrict__ W_l,    // [H, F, D]
                     const float* __restrict__ W_r,    // [H, F, D]
                     float*  __restrict__ g_l,         // [N, HD]
                     float*  __restrict__ g_r,         // [N, HD]
                     __bf16* __restrict__ g_r_bf)      // [N, HD]
{
    const int lane = threadIdx.x;            // 0..31
    const int m0   = blockIdx.x * 16;        // node-row tile
    const int head = blockIdx.y;             // 16-wide col tile == head (D==16)
    const bool is_r = (blockIdx.z == 1);
    const float* __restrict__ W = is_r ? W_r : W_l;

    __shared__ float w_lds[32 * 16];         // one 32(K) x 16(D) W tile
    const unsigned w_off = (unsigned)(uintptr_t)(&w_lds[0]);

    // A fragment addressing (16x32 bf16): lane -> row, element t -> K
    const int am    = lane & 15;
    const int ak_hi = (lane >> 4) << 4;      // +16 K-offset for upper half-wave
    // B fragment addressing (32x16 bf16): lane -> col, element t -> K
    const int bn    = lane & 15;
    const int bk0   = (lane >> 4) << 4;      // 0 or 16

    v8f acc = {};
    for (int k0 = 0; k0 < F_IN; k0 += 32) {
        // DMA the 32x16 fp32 W tile (rows contiguous, stride D) into LDS.
        tdm_load_2d(w_off, W + ((size_t)head * F_IN + k0) * D_HID,
                    /*4B*/2, /*d0*/D_HID, /*d1*/F_IN, /*tile*/D_HID, 32, D_HID);

        // Meanwhile gather + convert the A tile from h (two 8-float runs).
        v16bf a;
        const float* __restrict__ hrow = h + (size_t)(m0 + am) * F_IN + k0;
#pragma unroll
        for (int t = 0; t < 8; ++t) {
            a[t]     = (__bf16)hrow[ak_hi + t];           // K = ak_hi + t
            a[t + 8] = (__bf16)hrow[ak_hi + 8 + t];       // K = ak_hi + 8 + t
        }

        tdm_wait_all();
        v16bf b;
#pragma unroll
        for (int t = 0; t < 16; ++t)
            b[t] = (__bf16)w_lds[(bk0 + t) * D_HID + bn]; // K = bk0 + t
        acc = __builtin_amdgcn_wmma_f32_16x16x32_bf16(
                  false, a, false, b, (short)0, acc, false, false);
    }

    // C fragment store: row = 8*(lane>>4)+r, col = lane&15  (branch hoisted)
    const int cm = (lane >> 4) << 3;
    const int cn = lane & 15;
    float* __restrict__ gdst = is_r ? g_r : g_l;
#pragma unroll
    for (int r = 0; r < 8; ++r)
        gdst[(size_t)(m0 + cm + r) * HD + head * D_HID + cn] = acc[r];
    if (is_r) {
#pragma unroll
        for (int r = 0; r < 8; ++r)
            g_r_bf[(size_t)(m0 + cm + r) * HD + head * D_HID + cn] = (__bf16)acc[r];
    }
}

// ---------------------------------------------------------------------------
// Kernel 2: fused masked-softmax attention + aggregation, one wave per
// (16-row i-tile, head). Streams j in blocks of 32 (== WMMA K), keeping
// running row max m / denom s, accumulating P@V in a WMMA f32 fragment.
// V tiles arrive via TDM; next g_l block via global_prefetch.
// grid (N/16, H), block 32.
// ---------------------------------------------------------------------------
__global__ __launch_bounds__(32)
void gat_attn_kernel(const float*  __restrict__ g_l,     // [N, HD]
                     const float*  __restrict__ g_r,     // [N, HD]
                     const __bf16* __restrict__ g_r_bf,  // [N, HD]
                     const int*    __restrict__ adj,     // [N, N]
                     const float*  __restrict__ attn_w,  // [D]
                     float*        __restrict__ out)     // [N, HD]
{
    const int lane = threadIdx.x;
    const int i0   = blockIdx.x * 16;
    const int head = blockIdx.y;

    __shared__ float  gr_i[16][16];      // g_r rows of this i-tile (queries)
    __shared__ float  e_blk[16][32];     // raw scores for current j-block
    __shared__ __bf16 p_bf[16][32];      // exp(e - m) as bf16 (A fragment src)
    __shared__ __bf16 v_lds[32][16];     // V tile (g_r_bf j-block) via TDM
    __shared__ float  row_m[16], row_s[16], row_alpha[16];
    const unsigned v_off = (unsigned)(uintptr_t)(&v_lds[0][0]);

    float aw[16];
#pragma unroll
    for (int d = 0; d < 16; ++d) aw[d] = attn_w[d];

    for (int t = lane; t < 256; t += 32)
        gr_i[t >> 4][t & 15] =
            g_r[(size_t)(i0 + (t >> 4)) * HD + head * D_HID + (t & 15)];
    if (lane < 16) { row_m[lane] = NEG_BIG; row_s[lane] = 0.0f; }
    __syncthreads();

    const int am    = lane & 15;
    const int ak_hi = (lane >> 4) << 4;
    const int bn    = lane & 15;
    const int bk0   = (lane >> 4) << 4;
    const int crow  = (lane >> 4) << 3;

    v8f acc = {};
    for (int j0 = 0; j0 < N_NODES; j0 += 32) {
        // DMA this block's V tile (32 rows x 16 bf16, row stride HD) into LDS;
        // overlaps with the score computation below.
        tdm_load_2d(v_off, g_r_bf + (size_t)j0 * HD + head * D_HID,
                    /*2B*/1, /*d0*/D_HID, /*d1*/N_NODES, /*tile*/D_HID, 32, HD);

        // Prefetch next block's g_l rows while we compute this one.
        if (j0 + 32 < N_NODES)
            __builtin_prefetch(g_l + (size_t)(j0 + 32 + lane) * HD + head * D_HID, 0, 1);

        // ---- scores: lane owns column j = j0+lane, all 16 rows ----
        float glj[16];
        const float* __restrict__ glrow =
            g_l + (size_t)(j0 + lane) * HD + head * D_HID;
#pragma unroll
        for (int d = 0; d < 16; ++d) glj[d] = glrow[d];

#pragma unroll 4
        for (int i = 0; i < 16; ++i) {
            float e = 0.0f;
#pragma unroll
            for (int d = 0; d < 16; ++d) {
                const float x  = gr_i[i][d] + glj[d];
                const float lr = (x >= 0.0f) ? x : SLOPE * x;
                e = fmaf(aw[d], lr, e);
            }
            if (adj[(size_t)(i0 + i) * N_NODES + (j0 + lane)] == 0) e = NEG_BIG;
            e_blk[i][lane] = e;
        }
        __syncthreads();

        // ---- online softmax row pass: lanes 0..15 own one row each ----
        if (lane < 16) {
            const float m_old = row_m[lane];
            float bm = m_old;
#pragma unroll 8
            for (int j = 0; j < 32; ++j) bm = fmaxf(bm, e_blk[lane][j]);
            const float alpha = __expf(m_old - bm);
            float psum = 0.0f;
#pragma unroll 8
            for (int j = 0; j < 32; ++j) {
                const float p = __expf(e_blk[lane][j] - bm);
                p_bf[lane][j] = (__bf16)p;
                psum += p;
            }
            row_m[lane]     = bm;
            row_s[lane]     = row_s[lane] * alpha + psum;
            row_alpha[lane] = alpha;
        }
        __syncthreads();

        // ---- rescale accumulator fragment by per-row alpha ----
#pragma unroll
        for (int r = 0; r < 8; ++r) acc[r] *= row_alpha[crow + r];

        // ---- P (16x32 bf16, A frag) x V (32x16 bf16, B frag from TDM LDS) --
        v16bf a, b;
#pragma unroll
        for (int t = 0; t < 8; ++t) {
            a[t]     = p_bf[am][ak_hi + t];
            a[t + 8] = p_bf[am][ak_hi + 8 + t];
        }
        tdm_wait_all();
#pragma unroll
        for (int t = 0; t < 16; ++t)
            b[t] = v_lds[bk0 + t][bn];                    // K = bk0 + t
        acc = __builtin_amdgcn_wmma_f32_16x16x32_bf16(
                  false, a, false, b, (short)0, acc, false, false);
        __syncthreads();   // e_blk/p_bf/v_lds reads done before next iteration
    }

    // ---- normalize and store: out[i, head*16 + d] ----
#pragma unroll
    for (int r = 0; r < 8; ++r) {
        const int row   = crow + r;
        const float den = row_s[row];
        out[(size_t)(i0 + row) * HD + head * D_HID + (lane & 15)] = acc[r] / den;
    }
}

// ---------------------------------------------------------------------------
extern "C" void kernel_launch(void* const* d_in, const int* in_sizes, int n_in,
                              void* d_out, int out_size, void* d_ws, size_t ws_size,
                              hipStream_t stream) {
    const float* h      = (const float*)d_in[0];   // [1024,256]
    const int*   adj    = (const int*)  d_in[1];   // [1024,1024,1]
    const float* W_l    = (const float*)d_in[2];   // [8,256,16]
    const float* W_r    = (const float*)d_in[3];   // [8,256,16]
    const float* attn_w = (const float*)d_in[4];   // [16]
    float*       out    = (float*)d_out;           // [1024,128]

    char* ws = (char*)d_ws;
    float*  g_l    = (float*) (ws);                       // 512 KB
    float*  g_r    = (float*) (ws +  512 * 1024);         // 512 KB
    __bf16* g_r_bf = (__bf16*)(ws + 1024 * 1024);         // 256 KB

    dim3 g1(N_NODES / 16, H_HEADS, 2), b1(32);
    gat_proj_kernel<<<g1, b1, 0, stream>>>(h, W_l, W_r, g_l, g_r, g_r_bf);

    dim3 g2(N_NODES / 16, H_HEADS), b2(32);
    gat_attn_kernel<<<g2, b2, 0, stream>>>(g_l, g_r, g_r_bf, adj, attn_w, out);
}
